// TensorProductReference_65807488909770
// MI455X (gfx1250) — matrix operand order repaired and checked by reading, not compiled, not run on hardware
//
#include <hip/hip_runtime.h>

// ---------------------------------------------------------------------------
// CG tensor product  out[b,0:99] = W[99 x 256] . (x[b,:] (x) y[b,:])
// as a split-f16 WMMA GEMM on gfx1250 (v_wma_f32_16x16x32_f16, wave32).
// ---------------------------------------------------------------------------

typedef __attribute__((ext_vector_type(16))) _Float16 v16h;
typedef __attribute__((ext_vector_type(8)))  float    v8f;
typedef __attribute__((ext_vector_type(4)))  float    v4f;

#define NPATHS    23
#define W_COLS    112                 // 99 output cols padded to 7*16
#define NQ        256                 // 16x16 outer-product columns
#define WM_BYTES  (NQ * W_COLS * 4)   // 114688 B master W (f32)
#define NFRAGS    (7 * 8 * 2)         // 7 col tiles * 8 k-steps * {hi,lo}
#define FRAG_HALVES 512               // 32 lanes * 16 halves per fragment
#define NROWS     (4096 * 128)
#define NGROUPS   (NROWS / 16)

// ---- CG path tables (l1,l2,l3, output offset), parity-filtered, l3<=3 ----
__constant__ int PL1[NPATHS]  = {0,0,0,0, 1,1,1,1,1,1, 2,2,2,2,2,2,2, 3,3,3,3,3,3};
__constant__ int PL2[NPATHS]  = {0,1,2,3, 0,1,1,2,2,3, 0,1,1,2,2,3,3, 0,1,2,2,3,3};
__constant__ int PL3[NPATHS]  = {0,1,2,3, 1,0,2,1,3,2, 2,1,3,0,2,1,3, 3,2,1,3,0,2};
__constant__ int POFF[NPATHS] = {0,1,4,9, 16,19,20,25,28,35, 40,45,48,55,56,61,64, 71,78,83,86,93,94};
__constant__ double FACT[12] = {1.,1.,2.,6.,24.,120.,720.,5040.,40320.,362880.,3628800.,39916800.};

__device__ __forceinline__ int imax2(int a, int b) { return a > b ? a : b; }
__device__ __forceinline__ int imin2(int a, int b) { return a < b ? a : b; }

// Racah formula, literal translation of the reference _su2_cg (integer l,m).
__device__ double su2_cg(int j1, int j2, int j3, int m1, int m2, int m3) {
  if (m3 != m1 + m2) return 0.0;
  int vmin = imax2(imax2(-j1 + j2 + m3, -j1 + m1), 0);
  int vmax = imin2(imin2(j2 + j3 + m1, j3 - j1 + j2), j3 + m3);
  double C = sqrt((2.0 * j3 + 1.0) * FACT[j3 + j1 - j2] * FACT[j3 - j1 + j2] *
                  FACT[j1 + j2 - j3] * FACT[j3 + m3] * FACT[j3 - m3] /
                  (FACT[j1 + j2 + j3 + 1] * FACT[j1 - m1] * FACT[j1 + m1] *
                   FACT[j2 - m2] * FACT[j2 + m2]));
  double S = 0.0;
  for (int v = vmin; v <= vmax; ++v) {
    double sgn = ((v + j2 + m2) & 1) ? -1.0 : 1.0;
    S += sgn * FACT[j2 + j3 + m1 - v] * FACT[j1 - m1 + v] /
         (FACT[v] * FACT[j3 - j1 + j2 - v] * FACT[j3 + m3 - v] * FACT[v + j1 - j2 - m3]);
  }
  return C * S;
}

// Entry (a,b) of Q_l = (-i)^l * q_real_to_complex(l).
__device__ void qent(int l, int a, int b, double& re, double& im) {
  const double is2 = 0.70710678118654752440;
  double rr = 0.0, ri = 0.0;
  int m = a - l;
  if (m < 0) {
    if (b == l - m)      rr = is2;     // q[l+m, l+|m|] =  1/sqrt2
    else if (b == l + m) ri = -is2;    // q[l+m, l-|m|] = -i/sqrt2
  } else if (m == 0) {
    if (b == l) rr = 1.0;
  } else {
    double s = (m & 1) ? -1.0 : 1.0;
    if (b == l + m)      rr = s * is2;
    else if (b == l - m) ri = s * is2;
  }
  switch (l & 3) {                      // multiply by (-i)^l
    case 0: re =  rr; im =  ri; break;
    case 1: re =  ri; im = -rr; break;
    case 2: re = -rr; im = -ri; break;
    default: re = -ri; im =  rr; break;
  }
}

// ---- setup kernel A: zero master W ----------------------------------------
__global__ void zero_w(float* __restrict__ Wm) {
  int i = blockIdx.x * blockDim.x + threadIdx.x;
  if (i < NQ * W_COLS) Wm[i] = 0.0f;
}

// ---- setup kernel B: dense f32 master W[256][112] (one block) -------------
__global__ __launch_bounds__(256) void build_w(float* __restrict__ Wm) {
  __shared__ double Cr[343];
  __shared__ double s_inv;
  const int tid = threadIdx.x;
  for (int p = 0; p < NPATHS; ++p) {
    const int L1 = PL1[p], L2 = PL2[p], L3 = PL3[p];
    const int d1 = 2 * L1 + 1, d2 = 2 * L2 + 1, d3 = 2 * L3 + 1;
    const int nout = d1 * d2 * d3;
    for (int idx = tid; idx < nout; idx += blockDim.x) {
      const int m = idx % d3, l = (idx / d3) % d2, j = idx / (d3 * d2);
      double re = 0.0;
      for (int i = 0; i < d1; ++i)
        for (int k = 0; k < d2; ++k) {
          const int m1 = i - L1, m2 = k - L2, m3 = m1 + m2;
          if (m3 < -L3 || m3 > L3) continue;
          const double c = su2_cg(L1, L2, L3, m1, m2, m3);
          if (c == 0.0) continue;
          double ar, ai, br, bi, cr3, ci3;
          qent(L1, i, j, ar, ai);
          qent(L2, k, l, br, bi);
          qent(L3, L3 + m3, m, cr3, ci3);
          ci3 = -ci3;                                   // conj(Q3)
          const double pr = ar * br - ai * bi;
          const double pi = ar * bi + ai * br;
          re += (pr * cr3 - pi * ci3) * c;              // real part only
        }
      Cr[idx] = re;
    }
    __syncthreads();
    if (tid == 0) {
      double ss = 0.0;
      for (int i = 0; i < nout; ++i) ss += Cr[i] * Cr[i];
      s_inv = 1.0 / sqrt(ss);                           // unit Frobenius norm
    }
    __syncthreads();
    const int off1 = L1 * L1, off2 = L2 * L2, off3 = POFF[p];
    for (int idx = tid; idx < nout; idx += blockDim.x) {
      const int m = idx % d3, l = (idx / d3) % d2, j = idx / (d3 * d2);
      const int q = (off1 + j) * 16 + (off2 + l);
      Wm[q * W_COLS + (off3 + m)] = (float)(Cr[idx] * s_inv);
    }
    __syncthreads();
  }
}

// ---- setup kernel C: pre-swizzle W into f16 hi/lo B-fragments -------------
// B (32x16 f16, 8 VGPRs): element e=2v+h of lane half hh holds
// K = 16*(e>>3) + 8*hh + (e&7) (table order, matching the A fragment).
__global__ void build_frags(const float* __restrict__ Wm, unsigned int* __restrict__ fd) {
  const int idx = blockIdx.x * blockDim.x + threadIdx.x;   // fragidx*256 + lane*8 + v
  if (idx >= NFRAGS * 32 * 8) return;
  const int v = idx & 7, lane = (idx >> 3) & 31, fragidx = idx >> 8;
  const int part = fragidx & 1, s = (fragidx >> 1) & 7, t = fragidx >> 4;
  const int col = t * 16 + (lane & 15);
  unsigned int packed = 0;
  for (int h = 0; h < 2; ++h) {
    const int K = (v >> 2) * 16 + ((lane >> 4) & 1) * 8 + (v & 3) * 2 + h;
    const float w = Wm[(s * 32 + K) * W_COLS + col];
    _Float16 whi = (_Float16)w;
    _Float16 val = (part == 0) ? whi : (_Float16)(w - (float)whi);
    union { _Float16 f; unsigned short u; } cv; cv.f = val;
    packed |= (unsigned int)cv.u << (16 * h);
  }
  fd[idx] = packed;
}

// ---- main kernel: 1 wave = 16 batch rows, split-f16 WMMA GEMM -------------
__global__ __launch_bounds__(256)
void cg_tp_wmma(const float* __restrict__ x, const float* __restrict__ y,
                const _Float16* __restrict__ frags, float* __restrict__ out) {
  const int lane = threadIdx.x & 31;
  const int grp  = blockIdx.x * 8 + (threadIdx.x >> 5);
  const int row  = (grp << 4) + (lane & 15);      // both half-waves: same 16 rows

  const v4f* __restrict__ xp = (const v4f*)(x + (size_t)row * 16);
  const v4f* __restrict__ yp = (const v4f*)(y + (size_t)row * 16);
  float xr[16], yr[16];
#pragma unroll
  for (int i = 0; i < 4; ++i) {
    v4f a = xp[i], b = yp[i];
#pragma unroll
    for (int j = 0; j < 4; ++j) { xr[i * 4 + j] = a[j]; yr[i * 4 + j] = b[j]; }
  }
  const bool hih = (lane & 16) != 0;
  float ysel[8];                                  // j = (lane<16 ? c : c+8)
#pragma unroll
  for (int c = 0; c < 8; ++c) ysel[c] = hih ? yr[c + 8] : yr[c];

  v8f acc[7];
#pragma unroll
  for (int t = 0; t < 7; ++t) acc[t] = (v8f){0.f, 0.f, 0.f, 0.f, 0.f, 0.f, 0.f, 0.f};

#pragma unroll
  for (int s = 0; s < 8; ++s) {
    // A fragment (16x32 f16): element e of lane -> z[q], q = 32s + K with
    // i = 2s + (e>>3) (compile time), j = ysel[e&7]. Split z = z_hi + z_lo.
    v16h ah, al;
#pragma unroll
    for (int e = 0; e < 16; ++e) {
      const float z  = xr[2 * s + (e >> 3)] * ysel[e & 7];
      const _Float16 zh = (_Float16)z;
      ah[e] = zh;
      al[e] = (_Float16)(z - (float)zh);
    }
#pragma unroll
    for (int t = 0; t < 7; ++t) {
      const _Float16* fb = frags + ((t * 8 + s) * 2) * FRAG_HALVES + lane * 16;
      const v16h bh = *(const v16h*)fb;                 // W_hi fragment
      const v16h bl = *(const v16h*)(fb + FRAG_HALVES); // W_lo fragment
      acc[t] = __builtin_amdgcn_wmma_f32_16x16x32_f16(false, ah, false, bh,
                                                      (short)0, acc[t], false, false);
      acc[t] = __builtin_amdgcn_wmma_f32_16x16x32_f16(false, al, false, bh,
                                                      (short)0, acc[t], false, false);
      acc[t] = __builtin_amdgcn_wmma_f32_16x16x32_f16(false, ah, false, bl,
                                                      (short)0, acc[t], false, false);
    }
  }

  // D layout (f32 16x16): VGPR g -> M = g (lanes 0-15) / 8+g (lanes 16-31), N = lane%16
  const int n = lane & 15;
  const int mofs = hih ? 8 : 0;
#pragma unroll
  for (int t = 0; t < 7; ++t) {
    const int col = t * 16 + n;
    if (col < 99) {
#pragma unroll
      for (int g = 0; g < 8; ++g) {
        const size_t r = (size_t)(grp << 4) + mofs + g;
        __builtin_nontemporal_store(acc[t][g], &out[r * 99 + col]);
      }
    }
  }
}

extern "C" void kernel_launch(void* const* d_in, const int* in_sizes, int n_in,
                              void* d_out, int out_size, void* d_ws, size_t ws_size,
                              hipStream_t stream) {
  const float* x = (const float*)d_in[0];
  const float* y = (const float*)d_in[1];
  float* Wm            = (float*)d_ws;                         // 114688 B
  unsigned int* fdw    = (unsigned int*)((char*)d_ws + WM_BYTES);
  const _Float16* frag = (const _Float16*)((char*)d_ws + WM_BYTES);

  zero_w<<<(NQ * W_COLS + 255) / 256, 256, 0, stream>>>(Wm);
  build_w<<<1, 256, 0, stream>>>(Wm);
  build_frags<<<(NFRAGS * 32 * 8 + 255) / 256, 256, 0, stream>>>(Wm, fdw);
  cg_tp_wmma<<<NGROUPS / 8, 256, 0, stream>>>(x, y, frag, (float*)d_out);
}
